// DetectionGCN_39264591020294
// MI455X (gfx1250) — matrix operand order repaired and checked by reading, not compiled, not run on hardware
//
#include <hip/hip_runtime.h>
#include <hip/hip_bf16.h>

// ---------------------------------------------------------------------------
// 20-layer GCN on MI455X (gfx1250, wave32).
//   per layer:  H = h_in @ W          (bf16 WMMA, f32 accumulate)
//               OUT = H*dis^2 + b     (fused into GEMM epilogue)
//               OUT += scatter_dst( H[src] * norm )   (f32 global atomics, L2-resident)
//   ReLU folded into next layer's A-matrix load.
//   W tile staged via GLOBAL_LOAD_ASYNC_TO_LDS_B128 (ASYNCcnt path), then
//   converted f32 -> bf16 in LDS with packed v_cvt.
//   bf16 tile lives at LDS offset 0 so every hot-loop ds_load_b128 uses the
//   16-bit immediate offset field (no per-load address math).
// ---------------------------------------------------------------------------

typedef __attribute__((ext_vector_type(16))) __bf16         v16bf;
typedef __attribute__((ext_vector_type(8)))  __bf16         v8bf;
typedef __attribute__((ext_vector_type(16))) float          v16f;
typedef __attribute__((ext_vector_type(8)))  float          v8f;
typedef __attribute__((ext_vector_type(8)))  unsigned int   v8u;
typedef __attribute__((ext_vector_type(4)))  unsigned int   v4u;

// ------------------------------ prep kernels -------------------------------

__global__ void k_fill1(float* __restrict__ p, int n) {
    int i = blockIdx.x * blockDim.x + threadIdx.x;
    if (i < n) p[i] = 1.0f;                       // deg starts at 1 (self loop)
}

__global__ void k_degcnt(float* __restrict__ deg, const int* __restrict__ dst, int e) {
    int i = blockIdx.x * blockDim.x + threadIdx.x;
    if (i < e) atomicAdd(&deg[dst[i]], 1.0f);
}

__global__ void k_rsqrtk(float* __restrict__ p, int n) {
    int i = blockIdx.x * blockDim.x + threadIdx.x;
    if (i < n) p[i] = rsqrtf(p[i]);               // dis = deg^{-1/2}
}

__global__ void k_normk(float* __restrict__ nrm, const float* __restrict__ dis,
                        const int* __restrict__ src, const int* __restrict__ dst, int e) {
    int i = blockIdx.x * blockDim.x + threadIdx.x;
    if (i < e) nrm[i] = dis[src[i]] * dis[dst[i]];
}

// ------------------------------ GEMM (WMMA) --------------------------------
// A: [nrows x 128] f32 (optional ReLU on load). W: [128 x FOUT] f32.
// Writes H = A@W and OUT = H*dis^2 + bias (self-loop + bias init for scatter).

constexpr int WPITCH = 136;                 // bf16 elements per LDS row (272 B)
constexpr int WLDS_BYTES = 128 * WPITCH * 2;   // 34 816 B, 32B-aligned

template <int FOUT, bool RELU>
__global__ __launch_bounds__(256) void k_gcn_gemm(
    const float* __restrict__ A, const float* __restrict__ W,
    const float* __restrict__ bias, const float* __restrict__ dis,
    float* __restrict__ H, float* __restrict__ OUT, int nrows)
{
    constexpr int NELEM = 128 * FOUT;       // f32 elements of W
    // One carved LDS block: bf16 tile FIRST (offset 0 -> small DS immediates),
    // f32 async-staging region after it.
    __shared__ __align__(32) unsigned char smem[WLDS_BYTES + NELEM * 4];
    unsigned short* Wlds = (unsigned short*)smem;            // LDS offset 0
    float*          Wf32 = (float*)(smem + WLDS_BYTES);      // staging

    const int tid = threadIdx.x;

    // --- Stage 1: async DMA of W into LDS (no VGPR round trip, ASYNCcnt) ---
    for (int i = tid * 4; i < NELEM; i += 256 * 4) {
        unsigned           ldsa = (unsigned)(unsigned long long)(void*)&Wf32[i];
        unsigned long long ga   = (unsigned long long)(const void*)(W + i);
        asm volatile("global_load_async_to_lds_b128 %0, %1, off"
                     :: "v"(ldsa), "v"(ga) : "memory");
    }
    asm volatile("s_wait_asynccnt 0x0" ::: "memory");
    __syncthreads();

    // --- Stage 2: LDS f32 -> LDS bf16 (packed converts), pitched layout ---
    {
        constexpr int CPT = NELEM / 256;        // 64 (FOUT=128) or 32 (FOUT=64)
        const int base = tid * CPT;
#pragma unroll
        for (int off = 0; off < CPT; off += 8) {
            int idx = base + off;
            int k = idx / FOUT;
            int n = idx & (FOUT - 1);
            v8f  fv = *(const v8f*)&Wf32[idx];
            v8bf hv = __builtin_convertvector(fv, v8bf);
#pragma unroll
            for (int j = 0; j < 8; ++j)
                Wlds[(n + j) * WPITCH + k] = __builtin_bit_cast(unsigned short, hv[j]);
        }
    }
    __syncthreads();

    const int lane  = tid & 31;
    const int wave  = tid >> 5;
    const int m     = lane & 15;
    const int khalf = (lane & 16) ? 8 : 0;        // ISA 16-bit A 16x32 lane split
    const int row_base = blockIdx.x * 128 + wave * 16;
    const int arow  = row_base + m;
    const bool avalid = (arow < nrows);
    const float* ap = A + (size_t)arow * 128;

    // A fragments (K chunks of 32) per ISA layout:
    // lanes 0-15 (row m): K = {kc*32+0..7, kc*32+16..23}
    // lanes 16-31 (row m): K = {kc*32+8..15, kc*32+24..31}
    v16bf afrag[4];
#pragma unroll
    for (int kc = 0; kc < 4; ++kc) {
        v16f fv;
        if (avalid) {
            v8f lo = *(const v8f*)(ap + kc * 32 + khalf);
            v8f hi = *(const v8f*)(ap + kc * 32 + 16 + khalf);
            fv = __builtin_shufflevector(lo, hi,
                    0, 1, 2, 3, 4, 5, 6, 7, 8, 9, 10, 11, 12, 13, 14, 15);
        } else {
            fv = (v16f)0.0f;
        }
        if (RELU) fv = __builtin_elementwise_max(fv, (v16f)0.0f);
        afrag[kc] = __builtin_convertvector(fv, v16bf);
    }

    constexpr int NT = FOUT / 16;
    v8f acc[NT] = {};

    // B fragment: LDS holds W transposed (row n, contiguous k); lane holds column
    // N = nt*16 + (lane&15), K = {kc*32+khalf+0..7, kc*32+16+khalf+0..7}. kc/nt
    // displacements fold into the DS 16-bit immediate offset.
    const int krow_lo = (lane & 15);
    const unsigned short* wb = Wlds + krow_lo * WPITCH + khalf;
#pragma unroll
    for (int kc = 0; kc < 4; ++kc) {
#pragma unroll
        for (int nt = 0; nt < NT; ++nt) {
            v4u lo = *(const v4u*)(wb + nt * (16 * WPITCH) + kc * 32);
            v4u hi = *(const v4u*)(wb + nt * (16 * WPITCH) + kc * 32 + 16);
            v8u raw = __builtin_shufflevector(lo, hi, 0, 1, 2, 3, 4, 5, 6, 7);
            v16bf bfrag = __builtin_bit_cast(v16bf, raw);
            acc[nt] = __builtin_amdgcn_wmma_f32_16x16x32_bf16(
                false, afrag[kc], false, bfrag, (short)0, acc[nt], false, false);
        }
    }

    // Epilogue. C/D layout: lane -> col = nt*16 + (lane&15);
    // VGPR r -> row M = r (lanes 0-15) / 8+r (lanes 16-31).
    const int roff = (lane & 16) ? 8 : 0;
    float disv[8];
#pragma unroll
    for (int r = 0; r < 8; ++r) {
        int row = row_base + roff + r;
        disv[r] = (row < nrows) ? dis[row] : 0.0f;
    }
#pragma unroll
    for (int nt = 0; nt < NT; ++nt) {
        const int col = nt * 16 + m;
        const float bv = bias[col];
#pragma unroll
        for (int r = 0; r < 8; ++r) {
            int row = row_base + roff + r;
            if (row < nrows) {
                float v = acc[nt][r];
                size_t o = (size_t)row * FOUT + col;
                H[o]   = v;
                OUT[o] = v * disv[r] * disv[r] + bv;   // self-loop + bias
            }
        }
    }
}

// --------------------------- edge scatter (atomics) ------------------------
// One wave per edge; lane gathers F/32 contiguous floats of H[src], scales by
// norm[e], atomic-adds into OUT[dst]. Feature buffers (<=25.6MB) are
// L2-resident on MI455X (192MB L2), so this runs at L2 atomic throughput.

template <int F>
__global__ __launch_bounds__(256) void k_scatter(
    const float* __restrict__ H, float* __restrict__ OUT,
    const int* __restrict__ src, const int* __restrict__ dst,
    const float* __restrict__ nrm, int nE)
{
    int gid  = blockIdx.x * blockDim.x + threadIdx.x;
    int lane = threadIdx.x & 31;
    int e    = __builtin_amdgcn_readfirstlane(gid >> 5);  // wave-uniform -> SGPR
    if (e >= nE) return;

    const int   s = src[e];                               // scalar loads
    const int   d = dst[e];
    const float w = nrm[e];

    constexpr int PER = F / 32;   // 4 (F=128) or 2 (F=64)
    const float* hp = H   + (size_t)s * F + lane * PER;
    float*       op = OUT + (size_t)d * F + lane * PER;

    if constexpr (PER == 4) {
        float4 v = *(const float4*)hp;
        atomicAdd(op + 0, v.x * w);
        atomicAdd(op + 1, v.y * w);
        atomicAdd(op + 2, v.z * w);
        atomicAdd(op + 3, v.w * w);
    } else {
        float2 v = *(const float2*)hp;
        atomicAdd(op + 0, v.x * w);
        atomicAdd(op + 1, v.y * w);
    }
}

// ------------------------------- launcher ----------------------------------

extern "C" void kernel_launch(void* const* d_in, const int* in_sizes, int n_in,
                              void* d_out, int out_size, void* d_ws, size_t ws_size,
                              hipStream_t stream) {
    const float* x     = (const float*)d_in[0];
    const int*   ei    = (const int*)  d_in[1];
    const float* W_in  = (const float*)d_in[2];
    const float* b_in  = (const float*)d_in[3];
    const float* W_mid = (const float*)d_in[4];
    const float* b_mid = (const float*)d_in[5];
    const float* W_out = (const float*)d_in[6];
    const float* b_out = (const float*)d_in[7];

    const int nN   = in_sizes[0] / 128;          // 50000
    const int nE   = in_sizes[1] / 2;            // 800000
    const int nmid = in_sizes[4] / (128 * 128);  // 18

    const int* src = ei;
    const int* dst = ei + nE;

    // workspace layout (floats)
    float* dis  = (float*)d_ws;
    float* nrm  = dis + nN;
    float* H    = nrm + nE;
    float* bufA = H    + (size_t)nN * 128;
    float* bufB = bufA + (size_t)nN * 128;
    (void)ws_size;

    const int gN = (nN + 255) / 256;
    const int gE = (nE + 255) / 256;
    k_fill1 <<<gN, 256, 0, stream>>>(dis, nN);
    k_degcnt<<<gE, 256, 0, stream>>>(dis, dst, nE);
    k_rsqrtk<<<gN, 256, 0, stream>>>(dis, nN);
    k_normk <<<gE, 256, 0, stream>>>(nrm, dis, src, dst, nE);

    const int ggrid = (nN + 127) / 128;                         // 8 waves x 16 rows
    const int sgrid = (int)(((long long)nE * 32 + 255) / 256);  // 1 wave / edge

    // input layer (no ReLU on raw x)
    k_gcn_gemm<128, false><<<ggrid, 256, 0, stream>>>(x, W_in, b_in, dis, H, bufA, nN);
    k_scatter<128><<<sgrid, 256, 0, stream>>>(H, bufA, src, dst, nrm, nE);

    // hidden layers (ReLU of previous output folded into A load)
    float* cur = bufA;
    float* nxt = bufB;
    for (int l = 0; l < nmid; ++l) {
        k_gcn_gemm<128, true><<<ggrid, 256, 0, stream>>>(
            cur, W_mid + (size_t)l * 128 * 128, b_mid + (size_t)l * 128,
            dis, H, nxt, nN);
        k_scatter<128><<<sgrid, 256, 0, stream>>>(H, nxt, src, dst, nrm, nE);
        float* t = cur; cur = nxt; nxt = t;
    }

    // output layer: 128 -> 64, no final ReLU
    float* outp = (float*)d_out;
    k_gcn_gemm<64, true><<<ggrid, 256, 0, stream>>>(cur, W_out, b_out, dis, H, outp, nN);
    k_scatter<64><<<sgrid, 256, 0, stream>>>(H, outp, src, dst, nrm, nE);
}
